// GPT2LMHeadModel_56573309223278
// MI455X (gfx1250) — compile-verified, matrix operand order checked by reading, not run on hardware
//
#include <hip/hip_runtime.h>

// GPT-2 small forward for MI455X (gfx1250, wave32, WMMA + TDM).
// ~0.51 TFLOP of GEMM -> V_WMMA_F32_16X16X32_BF16. Weights are converted
// fp32->bf16 once (repeated tile re-reads then hit the 192 MB L2 as bf16);
// LayerNorm / attention / GELU emit bf16 activations, so the GEMM K-loop
// stages raw bf16 tiles with the Tensor Data Mover (double-buffered,
// s_wait_tensorcnt-pipelined) and spends its VALU/XDL budget on WMMA only.
// Attention is ~6 GF -> VALU online-softmax.

#define V_   50257
#define P_   1024
#define D_   768
#define L_   12
#define H_   12
#define HD_  64
#define DFF_ 3072
#define B_   2
#define S_   1024
#define M_   (B_ * S_)   // 2048 token rows
#define EPS_ 1e-5f

// Block tile 128x128x32; 8 wave32s in 4x2; each wave owns 32x64
// (2x4 WMMA tiles -> 8 v_wmma per K-step fed by 12 ds_load_b128).
#define BM 128
#define BN 128
#define BK 32

typedef __attribute__((ext_vector_type(16))) __bf16 v16bf;
typedef __attribute__((ext_vector_type(8)))  float  v8f;
typedef unsigned short ushort_t;

__device__ __forceinline__ unsigned short f2bf(float f) {
  union { float f; unsigned u; } v; v.f = f;
  unsigned r = v.u + 0x7FFFu + ((v.u >> 16) & 1u);  // round-to-nearest-even
  return (unsigned short)(r >> 16);
}
__device__ __forceinline__ unsigned pk2bf(float a, float b) {
  return (unsigned)f2bf(a) | ((unsigned)f2bf(b) << 16);
}

// ------------------------------------------------------------- TDM support
#if __has_builtin(__builtin_amdgcn_tensor_load_to_lds) && \
    __has_builtin(__builtin_amdgcn_s_wait_tensorcnt)
#define HAVE_TDM 1
typedef unsigned __attribute__((ext_vector_type(4))) u32x4_t;
typedef int      __attribute__((ext_vector_type(8))) i32x8_t;
typedef int      __attribute__((ext_vector_type(4))) i32x4_t;

// Issue one 2D bf16 tile load (tile_dim0=32 cols, tile_dim1=128 rows) from a
// row-major [tdim1 x tdim0] tensor slice into LDS. Rows past tdim1 zero-fill
// (hardware OOB), which handles the ragged LM-head N=50257 edge for free.
__device__ __forceinline__ void tdm_load_tile(
    unsigned lds_off, const ushort_t* gbase, unsigned long long elem_off,
    unsigned tdim0, unsigned tdim1, unsigned stride) {
  unsigned long long ga = (unsigned long long)(uintptr_t)gbase + 2ull * elem_off;
  u32x4_t g0;
  g0[0] = 1u;                                         // count=1, user mode
  g0[1] = lds_off;                                    // LDS byte address
  g0[2] = (unsigned)ga;                               // global_addr[31:0]
  g0[3] = ((unsigned)(ga >> 32) & 0x1FFFFFFu) | (2u << 30);  // [56:32]|type=2
  union { unsigned w[8]; i32x8_t v; } g1;
  g1.w[0] = 1u << 16;                                 // data_size=1 -> 2 bytes
  g1.w[1] = (tdim0 & 0xFFFFu) << 16;                  // tensor_dim0[15:0]
  g1.w[2] = (tdim0 >> 16) | ((tdim1 & 0xFFFFu) << 16);// dim0[31:16]|dim1[15:0]
  g1.w[3] = (tdim1 >> 16) | ((unsigned)BK << 16);     // dim1[31:16]|tile_dim0
  g1.w[4] = (unsigned)BM;                             // tile_dim1; tile_dim2=0
  g1.w[5] = stride;                                   // tensor_dim0_stride lo
  g1.w[6] = 0;
  g1.w[7] = 0;                                        // tensor_dim1_stride=0
  i32x4_t z4 = {0, 0, 0, 0};
  i32x8_t z8 = {0, 0, 0, 0, 0, 0, 0, 0};
  // 6-arg form (this toolchain): (g0, g1, g2, g3, <reserved group>, cpol).
  __builtin_amdgcn_tensor_load_to_lds(g0, g1.v, z4, z4, z8, 0);
}
#else
#define HAVE_TDM 0
#endif

// -------------------------------------------------- fp32 -> bf16 converter
__global__ __launch_bounds__(256) void cvt_bf16_kernel(
    const float* __restrict__ src, ushort_t* __restrict__ dst, int n4) {
  int i = blockIdx.x * blockDim.x + threadIdx.x;
  if (i >= n4) return;
  float4 v = ((const float4*)src)[i];
  ((uint2*)dst)[i] = make_uint2(pk2bf(v.x, v.y), pk2bf(v.z, v.w));
}

// ---------------------------------------------------------------- embedding
__global__ __launch_bounds__(256) void embed_kernel(
    const int* __restrict__ ids, const float* __restrict__ wte,
    const float* __restrict__ wpe, float* __restrict__ h) {
  int i = blockIdx.x * blockDim.x + threadIdx.x;
  if (i >= M_ * D_) return;
  int d = i % D_;
  int t = i / D_;
  int s = t % S_;
  int id = ids[t];
  h[i] = wte[(size_t)id * D_ + d] + wpe[(size_t)s * D_ + d];
}

// ------------------------------------------- layernorm: fp32 in, bf16 out
__global__ __launch_bounds__(256) void ln_kernel(
    const float* __restrict__ x, const float* __restrict__ g,
    const float* __restrict__ b, ushort_t* __restrict__ y) {
  __shared__ float red[256];
  int row = blockIdx.x;
  const float* xr = x + (size_t)row * D_;
  float s = 0.f;
  for (int d = threadIdx.x; d < D_; d += 256) s += xr[d];
  red[threadIdx.x] = s;
  __syncthreads();
  for (int o = 128; o > 0; o >>= 1) {
    if ((int)threadIdx.x < o) red[threadIdx.x] += red[threadIdx.x + o];
    __syncthreads();
  }
  float mu = red[0] * (1.0f / D_);
  __syncthreads();
  float var = 0.f;
  for (int d = threadIdx.x; d < D_; d += 256) {
    float t = xr[d] - mu; var += t * t;
  }
  red[threadIdx.x] = var;
  __syncthreads();
  for (int o = 128; o > 0; o >>= 1) {
    if ((int)threadIdx.x < o) red[threadIdx.x] += red[threadIdx.x + o];
    __syncthreads();
  }
  float rstd = rsqrtf(red[0] * (1.0f / D_) + EPS_);
  for (int d = threadIdx.x; d < D_; d += 256)
    y[(size_t)row * D_ + d] = f2bf((xr[d] - mu) * rstd * g[d] + b[d]);
}

// ------------------------------------------------------- WMMA bf16 GEMM
// C[M,N] = act(A[M,K] @ W[N,K]^T + bias) (+ resid); A,W bf16; C fp32 or bf16.
__device__ __forceinline__ void wmma_step(
    const ushort_t* sa, const ushort_t* sw,
    int wm, int wn, int half, int l16, v8f acc[2][4]) {
  // Fragment gather per CDNA5 16-bit 16x32 layout: VGPR v holds the K-pair
  // at ((v>>2)<<4) + (half<<3) + ((v&3)<<1)  -> two ds_load_b128 each.
  union Frag { v16bf v; unsigned u[8]; };
  Frag af[2], bfr[4];
#pragma unroll
  for (int t = 0; t < 2; t++) {
    int arow = (wm * 32 + t * 16 + l16) * BK;
#pragma unroll
    for (int vv = 0; vv < 8; vv++) {
      int kk = ((vv >> 2) << 4) + (half << 3) + ((vv & 3) << 1);
      af[t].u[vv] = *(const unsigned*)&sa[arow + kk];
    }
  }
#pragma unroll
  for (int t = 0; t < 4; t++) {
    int brow = (wn * 64 + t * 16 + l16) * BK;
#pragma unroll
    for (int vv = 0; vv < 8; vv++) {
      int kk = ((vv >> 2) << 4) + (half << 3) + ((vv & 3) << 1);
      bfr[t].u[vv] = *(const unsigned*)&sw[brow + kk];
    }
  }
#pragma unroll
  for (int i = 0; i < 2; i++)
#pragma unroll
    for (int j = 0; j < 4; j++)
      acc[i][j] = __builtin_amdgcn_wmma_f32_16x16x32_bf16(
          false, af[i].v, false, bfr[j].v, (short)0, acc[i][j], false, false);
}

__global__ __launch_bounds__(256) void gemm_bf16_kernel(
    const ushort_t* __restrict__ A, const ushort_t* __restrict__ W,
    const float* __restrict__ bias, const float* __restrict__ resid,
    void* __restrict__ Cout, int Nn, int Kk, int act, int outbf) {
  __shared__ ushort_t sA[2][BM * BK];   // 2 x 8 KB (double buffer)
  __shared__ ushort_t sW[2][BN * BK];   // 2 x 8 KB

  const int tid  = threadIdx.x;
  const int lane = tid & 31;
  const int wave = tid >> 5;
  const int wm   = wave >> 1;
  const int wn   = wave & 1;
  const int m0   = blockIdx.y * BM;
  const int n0   = blockIdx.x * BN;
  const int half = lane >> 4;
  const int l16  = lane & 15;

  v8f acc[2][4];
  const v8f zacc = {0.f, 0.f, 0.f, 0.f, 0.f, 0.f, 0.f, 0.f};
#pragma unroll
  for (int i = 0; i < 2; i++)
#pragma unroll
    for (int j = 0; j < 4; j++) acc[i][j] = zacc;

  const int nsteps = Kk / BK;

#if HAVE_TDM
  // TDM-staged, double-buffered pipeline. Wave 0 drives the DMA; tensor ops
  // from one wave complete in order, so s_wait_tensorcnt(2) after issuing
  // the next tile's pair guarantees the current tile's pair has landed.
  const unsigned ldsA[2] = {(unsigned)(uintptr_t)&sA[0][0],
                            (unsigned)(uintptr_t)&sA[1][0]};
  const unsigned ldsW[2] = {(unsigned)(uintptr_t)&sW[0][0],
                            (unsigned)(uintptr_t)&sW[1][0]};
  const unsigned rowsA = (unsigned)(M_ - m0);
  const unsigned rowsW = (unsigned)(Nn - n0);   // ragged edge -> HW zero-fill
  if (wave == 0) {
    tdm_load_tile(ldsA[0], A, (unsigned long long)m0 * Kk, Kk, rowsA, Kk);
    tdm_load_tile(ldsW[0], W, (unsigned long long)n0 * Kk, Kk, rowsW, Kk);
  }
  for (int s = 0; s < nsteps; ++s) {
    const int cur = s & 1, nxt = cur ^ 1;
    if (wave == 0) {
      if (s + 1 < nsteps) {
        unsigned long long k1 = (unsigned long long)(s + 1) * BK;
        tdm_load_tile(ldsA[nxt], A, (unsigned long long)m0 * Kk + k1,
                      Kk, rowsA, Kk);
        tdm_load_tile(ldsW[nxt], W, (unsigned long long)n0 * Kk + k1,
                      Kk, rowsW, Kk);
        __builtin_amdgcn_s_wait_tensorcnt(2);
      } else {
        __builtin_amdgcn_s_wait_tensorcnt(0);
      }
    }
    __syncthreads();
    wmma_step(sA[cur], sW[cur], wm, wn, half, l16, acc);
    __syncthreads();
  }
#else
  // Fallback: per-thread vectorized staging (uint4 = 8 bf16 per load).
  for (int s = 0; s < nsteps; ++s) {
    int k0 = s * BK;
    const bool pf = (s + 1) < nsteps;
#pragma unroll
    for (int i = 0; i < 2; i++) {          // A tile: 512 chunks / 256 thr
      int c = tid + i * 256;
      int r = c >> 2, c8 = (c & 3) << 3;
      const ushort_t* ap = A + (size_t)(m0 + r) * Kk + k0 + c8;
      if (pf) __builtin_prefetch(ap + BK, 0, 3);
      *(uint4*)&sA[0][r * BK + c8] = *(const uint4*)ap;
    }
#pragma unroll
    for (int i = 0; i < 2; i++) {          // W tile
      int c = tid + i * 256;
      int r = c >> 2, c8 = (c & 3) << 3;
      int n = n0 + r;
      uint4 wv = make_uint4(0u, 0u, 0u, 0u);
      if (n < Nn) {
        const ushort_t* wp = W + (size_t)n * Kk + k0 + c8;
        wv = *(const uint4*)wp;
        if (pf) __builtin_prefetch(wp + BK, 0, 3);
      }
      *(uint4*)&sW[0][r * BK + c8] = wv;
    }
    __syncthreads();
    wmma_step(sA[0], sW[0], wm, wn, half, l16, acc);
    __syncthreads();
  }
#endif

  // Epilogue per C/D layout: VGPR r -> M = r + 8*half, N = lane%16.
#pragma unroll
  for (int i = 0; i < 2; i++) {
#pragma unroll
    for (int j = 0; j < 4; j++) {
#pragma unroll
      for (int r = 0; r < 8; r++) {
        int row = m0 + wm * 32 + i * 16 + half * 8 + r;
        int col = n0 + wn * 64 + j * 16 + l16;
        if (col < Nn) {
          float v = acc[i][j][r];
          if (bias)  v += bias[col];
          if (act)   v = 0.5f * v * (1.0f + erff(v * 0.70710678118654752f));
          if (resid) v += resid[(size_t)row * Nn + col];
          if (outbf) ((ushort_t*)Cout)[(size_t)row * Nn + col] = f2bf(v);
          else       ((float*)Cout)[(size_t)row * Nn + col] = v;
        }
      }
    }
  }
}

// --------------------------------------------------- causal attention (VALU)
// One thread per (b, h, q): online softmax over keys 0..q; bf16 output.
__global__ __launch_bounds__(256) void attn_kernel(
    const float* __restrict__ qkv, ushort_t* __restrict__ o) {
  int t = blockIdx.x * blockDim.x + threadIdx.x;
  if (t >= B_ * H_ * S_) return;
  int q = t & (S_ - 1);
  int h = (t >> 10) % H_;
  int b = t / (H_ * S_);

  const float* base = qkv + (size_t)(b * S_) * 3 * D_;
  const float* qp = base + (size_t)q * 3 * D_ + h * HD_;

  float qv[HD_], ov[HD_];
#pragma unroll
  for (int d = 0; d < HD_; d++) { qv[d] = qp[d]; ov[d] = 0.f; }

  float mrun = -3.4e38f, lrun = 0.f;
  for (int k = 0; k <= q; k++) {
    const float* kp = base + (size_t)k * 3 * D_ + D_ + h * HD_;
    float s = 0.f;
#pragma unroll
    for (int d = 0; d < HD_; d++) s += qv[d] * kp[d];
    s *= 0.125f;                          // 1/sqrt(64)
    float mn    = fmaxf(mrun, s);
    float alpha = __expf(mrun - mn);
    float p     = __expf(s - mn);
    lrun = lrun * alpha + p;
    const float* vp = base + (size_t)k * 3 * D_ + 2 * D_ + h * HD_;
#pragma unroll
    for (int d = 0; d < HD_; d++) ov[d] = ov[d] * alpha + p * vp[d];
    mrun = mn;
  }
  float inv = 1.f / lrun;
  ushort_t* op = o + (size_t)(b * S_ + q) * D_ + h * HD_;
#pragma unroll
  for (int d = 0; d < HD_; d++) op[d] = f2bf(ov[d] * inv);
}

// --------------------------------------------------------------- launcher
extern "C" void kernel_launch(void* const* d_in, const int* in_sizes, int n_in,
                              void* d_out, int out_size, void* d_ws,
                              size_t ws_size, hipStream_t stream) {
  (void)in_sizes; (void)n_in; (void)out_size; (void)ws_size;
  const int*   ids   = (const int*)  d_in[0];
  const float* wte   = (const float*)d_in[1];
  const float* wpe   = (const float*)d_in[2];
  const float* ln1g  = (const float*)d_in[3];
  const float* ln1b  = (const float*)d_in[4];
  const float* qkvw  = (const float*)d_in[5];
  const float* qkvb  = (const float*)d_in[6];
  const float* projw = (const float*)d_in[7];
  const float* projb = (const float*)d_in[8];
  const float* ln2g  = (const float*)d_in[9];
  const float* ln2b  = (const float*)d_in[10];
  const float* fcw   = (const float*)d_in[11];
  const float* fcb   = (const float*)d_in[12];
  const float* cpw   = (const float*)d_in[13];
  const float* cpb   = (const float*)d_in[14];
  const float* lnfg  = (const float*)d_in[15];
  const float* lnfb  = (const float*)d_in[16];

  // Workspace: bf16 weights (~247 MB) + activations (~44 MB).
  char* ws = (char*)d_ws;
  ushort_t* wqkv = (ushort_t*)ws; ws += (size_t)L_ * 3 * D_ * D_ * 2;
  ushort_t* wprj = (ushort_t*)ws; ws += (size_t)L_ * D_ * D_ * 2;
  ushort_t* wfc  = (ushort_t*)ws; ws += (size_t)L_ * DFF_ * D_ * 2;
  ushort_t* wcp  = (ushort_t*)ws; ws += (size_t)L_ * D_ * DFF_ * 2;
  ushort_t* wteb = (ushort_t*)ws; ws += (size_t)V_ * D_ * 2;
  float*    h    = (float*)ws;    ws += (size_t)M_ * D_ * 4;
  float*    qkv  = (float*)ws;    ws += (size_t)M_ * 3 * D_ * 4;
  ushort_t* hn   = (ushort_t*)ws; ws += (size_t)M_ * D_ * 2;
  ushort_t* ob   = (ushort_t*)ws; ws += (size_t)M_ * D_ * 2;
  ushort_t* mb   = (ushort_t*)ws; ws += (size_t)M_ * DFF_ * 2;

  dim3 blk(256);
  // One-time weight conversion (each weight is re-read per M-tile in the
  // GEMMs; bf16 halves that traffic and keeps per-layer tiles L2-resident).
  {
    struct { const float* s; ushort_t* d; size_t n; } cv[5] = {
      { qkvw,  wqkv, (size_t)L_ * 3 * D_ * D_ },
      { projw, wprj, (size_t)L_ * D_ * D_ },
      { fcw,   wfc,  (size_t)L_ * DFF_ * D_ },
      { cpw,   wcp,  (size_t)L_ * D_ * DFF_ },
      { wte,   wteb, (size_t)V_ * D_ },
    };
    for (int i = 0; i < 5; i++) {
      int n4 = (int)(cv[i].n / 4);
      cvt_bf16_kernel<<<dim3((n4 + 255) / 256), blk, 0, stream>>>(
          cv[i].s, cv[i].d, n4);
    }
  }

  embed_kernel<<<dim3((M_ * D_ + 255) / 256), blk, 0, stream>>>(ids, wte, wpe, h);

  for (int l = 0; l < L_; l++) {
    ln_kernel<<<dim3(M_), blk, 0, stream>>>(h, ln1g + l * D_, ln1b + l * D_, hn);
    gemm_bf16_kernel<<<dim3((3 * D_) / BN, M_ / BM), blk, 0, stream>>>(
        hn, wqkv + (size_t)l * 3 * D_ * D_, qkvb + (size_t)l * 3 * D_,
        nullptr, qkv, 3 * D_, D_, 0, 0);
    attn_kernel<<<dim3((B_ * H_ * S_) / 256), blk, 0, stream>>>(qkv, ob);
    gemm_bf16_kernel<<<dim3(D_ / BN, M_ / BM), blk, 0, stream>>>(
        ob, wprj + (size_t)l * D_ * D_, projb + (size_t)l * D_,
        h, h, D_, D_, 0, 0);
    ln_kernel<<<dim3(M_), blk, 0, stream>>>(h, ln2g + l * D_, ln2b + l * D_, hn);
    gemm_bf16_kernel<<<dim3(DFF_ / BN, M_ / BM), blk, 0, stream>>>(
        hn, wfc + (size_t)l * DFF_ * D_, fcb + (size_t)l * DFF_,
        nullptr, mb, DFF_, D_, 1, 1);
    gemm_bf16_kernel<<<dim3(D_ / BN, M_ / BM), blk, 0, stream>>>(
        mb, wcp + (size_t)l * D_ * DFF_, cpb + (size_t)l * D_,
        h, h, D_, DFF_, 0, 0);
  }

  ln_kernel<<<dim3(M_), blk, 0, stream>>>(h, lnfg, lnfb, hn);
  // Tied LM head: logits = hn @ wte^T; ragged N handled by TDM OOB zero-fill
  // (and epilogue bounds checks).
  gemm_bf16_kernel<<<dim3((V_ + BN - 1) / BN, M_ / BM), blk, 0, stream>>>(
      hn, wteb, nullptr, nullptr, (float*)d_out, V_, D_, 0, 0);
}
